// Block_40742059770386
// MI455X (gfx1250) — matrix operand order repaired and checked by reading, not compiled, not run on hardware
//
#include <hip/hip_runtime.h>
#include <hip/hip_bf16.h>

typedef __attribute__((ext_vector_type(8)))  _Float16 half8;
typedef __attribute__((ext_vector_type(16))) _Float16 v16h;
typedef __attribute__((ext_vector_type(8)))  float    v8f;

#define BATCH 16
#define C_IN  256
#define C_OUT 256
#define HH    64
#define WW    64
#define WPAD  66   // padded row width: zero pixel at col 0 and col 65

__device__ __forceinline__ v16h cat16(half8 lo, half8 hi) {
    return __builtin_shufflevector(lo, hi, 0,1,2,3,4,5,6,7,8,9,10,11,12,13,14,15);
}

// ---------------------------------------------------------------------------
// Kernel 1a: x [B][C][H][W] f32 -> x_t [B][H][WPAD][C] f16 (NHWC, width-padded)
// pixel w lands at padded column w+1; columns 0 and 65 are zero borders.
// ---------------------------------------------------------------------------
__global__ __launch_bounds__(256) void prep_x(const float* __restrict__ x,
                                              _Float16* __restrict__ xt) {
    const int c   = threadIdx.x;           // 0..255
    const int pix = blockIdx.x;            // (b*H + h)*W + w
    const int b   = pix / (HH * WW);
    const int hw  = pix % (HH * WW);
    const int h   = hw / WW;
    const int w   = hw % WW;
    float v = x[(size_t)(b * C_IN + c) * (HH * WW) + hw];
    xt[((size_t)(b * HH + h) * WPAD + (w + 1)) * C_IN + c] = (_Float16)v;
}

// Kernel 1b: zero the left/right border columns of every padded row.
__global__ __launch_bounds__(256) void pad_x(_Float16* __restrict__ xt) {
    const int c  = threadIdx.x;
    const int bh = blockIdx.x;             // b*H + h
    xt[((size_t)bh * WPAD + 0)         * C_IN + c] = (_Float16)0.0f;
    xt[((size_t)bh * WPAD + (WPAD - 1)) * C_IN + c] = (_Float16)0.0f;
}

// ---------------------------------------------------------------------------
// Kernel 2: softmax-mix kernel bank, modulate, demodulate; store f16 weights
// as w_t [B][9][C_IN][C_OUT]  (C_OUT contiguous -> B-fragment rows)
// ---------------------------------------------------------------------------
__global__ __launch_bounds__(256) void prep_w(const float* __restrict__ mod,
                                              const float* __restrict__ kmod,
                                              const float* __restrict__ wbank,
                                              _Float16* __restrict__ wt) {
    const int i = threadIdx.x;             // input channel
    const int o = blockIdx.x;              // output channel
    const int b = blockIdx.y;

    const float k0 = kmod[b * 2 + 0], k1 = kmod[b * 2 + 1];
    const float mx = fmaxf(k0, k1);
    const float e0 = __expf(k0 - mx), e1 = __expf(k1 - mx);
    const float inv = 1.0f / (e0 + e1);
    const float a0 = e0 * inv, a1 = e1 * inv;
    const float m1 = mod[b * C_IN + i] + 1.0f;

    const float* w0p = wbank + ((size_t)(0 * C_OUT + o) * C_IN + i) * 9;
    const float* w1p = wbank + ((size_t)(1 * C_OUT + o) * C_IN + i) * 9;

    float wreg[9];
    float ss = 0.0f;
#pragma unroll
    for (int kl = 0; kl < 9; ++kl) {
        float wv = (a0 * w0p[kl] + a1 * w1p[kl]) * m1;
        wreg[kl] = wv;
        ss += wv * wv;
    }

    __shared__ float red[256];
    red[i] = ss;
    __syncthreads();
#pragma unroll
    for (int s = 128; s > 0; s >>= 1) {
        if (i < s) red[i] += red[i + s];
        __syncthreads();
    }
    const float scale = rsqrtf(fmaxf(red[0], 1e-8f));

#pragma unroll
    for (int kl = 0; kl < 9; ++kl) {
        wt[((size_t)(b * 9 + kl) * C_IN + i) * C_OUT + o] =
            (_Float16)(wreg[kl] * scale);
    }
}

// ---------------------------------------------------------------------------
// Kernel 3: per-sample 3x3 conv as 9 shifted 1x1 GEMMs via WMMA f16->f32.
// Block = (b, row h, 32-pixel strip). 8 waves, each wave: 32 px x 32 couts
// via 2x2 register blocking (2 A-fragments x 2 B-fragments -> 4 WMMAs/step).
// Zero-padded x_t removes all per-lane boundary predication: EXEC stays
// all-ones through the WMMA loop. Writes NCHW f32 y.
// ---------------------------------------------------------------------------
__global__ __launch_bounds__(256) void conv_wmma(const _Float16* __restrict__ xt,
                                                 const _Float16* __restrict__ wt,
                                                 float* __restrict__ y) {
    const int lane  = threadIdx.x & 31;
    const int wave  = threadIdx.x >> 5;
    const int w0    = blockIdx.x * 32;     // 32-pixel strip
    const int h     = blockIdx.y;
    const int b     = blockIdx.z;
    const int o0    = wave * 32;           // 8 waves cover all 256 couts

    const int m     = lane & 15;           // A-matrix row (pixel within tile)
    const int khalf = lane >> 4;           // 0: K 0-7 & 16-23, 1: K 8-15 & 24-31

    v8f acc00 = {}, acc01 = {}, acc10 = {}, acc11 = {};

    for (int dk = 0; dk < 3; ++dk) {
        const int hs = h + dk - 1;
        if (hs < 0 || hs >= HH) continue;  // uniform skip: tap contributes 0
        const _Float16* rowbase = xt + (size_t)(b * HH + hs) * WPAD * C_IN;
        for (int dl = 0; dl < 3; ++dl) {
            // padded column of pixel (w0+m) shifted by dl-1 is (w0+m)+dl
            const _Float16* xp0 = rowbase + (size_t)(w0 + m + dl) * C_IN;
            const _Float16* xp1 = xp0 + (size_t)16 * C_IN;
            const _Float16* wtap =
                wt + (size_t)(b * 9 + dk * 3 + dl) * C_IN * C_OUT;

#pragma unroll 2
            for (int kc = 0; kc < C_IN; kc += 32) {
                const int ko = kc + khalf * 8;
                // A fragments: two 16-pixel tiles x 32 input channels
                v16h a0 = cat16(*(const half8*)(xp0 + ko),
                                *(const half8*)(xp0 + ko + 16));
                v16h a1 = cat16(*(const half8*)(xp1 + ko),
                                *(const half8*)(xp1 + ko + 16));
                // B fragments: 32 input channels x two 16-cout tiles
                const _Float16* wr = wtap + (size_t)ko * C_OUT + o0 + m;
                v16h b0, b1;
#pragma unroll
                for (int s = 0; s < 8; ++s) {
                    b0[s]     = wr[(size_t)s * C_OUT];
                    b0[s + 8] = wr[(size_t)(s + 16) * C_OUT];
                    b1[s]     = wr[(size_t)s * C_OUT + 16];
                    b1[s + 8] = wr[(size_t)(s + 16) * C_OUT + 16];
                }

                acc00 = __builtin_amdgcn_wmma_f32_16x16x32_f16(
                    false, a0, false, b0, (short)0, acc00, false, false);
                acc01 = __builtin_amdgcn_wmma_f32_16x16x32_f16(
                    false, a0, false, b1, (short)0, acc01, false, false);
                acc10 = __builtin_amdgcn_wmma_f32_16x16x32_f16(
                    false, a1, false, b0, (short)0, acc10, false, false);
                acc11 = __builtin_amdgcn_wmma_f32_16x16x32_f16(
                    false, a1, false, b1, (short)0, acc11, false, false);
            }
        }
    }

    // D layout: vgpr r -> M = r (lanes 0-15) / r+8 (lanes 16-31); N = lane&15
    const int n    = lane & 15;
    const int mrow = khalf * 8;
#pragma unroll
    for (int r = 0; r < 8; ++r) {
        const int wp0 = w0 + mrow + r;       // pixel tile 0
        const int wp1 = wp0 + 16;            // pixel tile 1
        y[((size_t)(b * C_OUT + o0 + n)      * HH + h) * WW + wp0] = acc00[r];
        y[((size_t)(b * C_OUT + o0 + 16 + n) * HH + h) * WW + wp0] = acc01[r];
        y[((size_t)(b * C_OUT + o0 + n)      * HH + h) * WW + wp1] = acc10[r];
        y[((size_t)(b * C_OUT + o0 + 16 + n) * HH + h) * WW + wp1] = acc11[r];
    }
}

// ---------------------------------------------------------------------------
// Kernel 4: in-place channel RMS-norm (* gamma * sqrt(C_OUT)) + SiLU.
// Block = (b, h) row: 64 pixels x 256 channels; coalesced channel-group loops.
// ---------------------------------------------------------------------------
__global__ __launch_bounds__(256) void norm_silu(float* __restrict__ y,
                                                 const float* __restrict__ gamma) {
    const int t  = threadIdx.x;
    const int h  = blockIdx.x;
    const int b  = blockIdx.y;
    const int w  = t & 63;
    const int cg = t >> 6;                 // channel group 0..3 (64 channels each)

    __shared__ float sums[4][64];

    float ss = 0.0f;
    for (int c = cg * 64; c < (cg + 1) * 64; ++c) {
        float v = y[((size_t)(b * C_OUT + c) * HH + h) * WW + w];
        ss += v * v;
    }
    sums[cg][w] = ss;
    __syncthreads();

    const float tot   = sums[0][w] + sums[1][w] + sums[2][w] + sums[3][w];
    const float scale = 16.0f / fmaxf(sqrtf(tot), 1e-12f);   // sqrt(256) = 16

    for (int c = cg * 64; c < (cg + 1) * 64; ++c) {
        const size_t idx = ((size_t)(b * C_OUT + c) * HH + h) * WW + w;
        float v = y[idx] * scale * gamma[c];
        y[idx]  = v / (1.0f + __expf(-v));   // SiLU
    }
}

// ---------------------------------------------------------------------------
extern "C" void kernel_launch(void* const* d_in, const int* in_sizes, int n_in,
                              void* d_out, int out_size, void* d_ws, size_t ws_size,
                              hipStream_t stream) {
    const float* x       = (const float*)d_in[0];   // [16,256,64,64]
    const float* mod     = (const float*)d_in[1];   // [16,256]
    const float* kmod    = (const float*)d_in[2];   // [16,2]
    const float* weights = (const float*)d_in[3];   // [2,256,256,3,3]
    const float* gamma   = (const float*)d_in[4];   // [256]
    float*       out     = (float*)d_out;           // [16,256,64,64]

    const size_t xt_bytes = (size_t)BATCH * HH * WPAD * C_IN * sizeof(_Float16); // ~34.6 MB
    _Float16* xt = (_Float16*)d_ws;
    _Float16* wt = (_Float16*)((char*)d_ws + xt_bytes);                          // ~18.9 MB

    prep_x   <<<dim3(BATCH * HH * WW), 256, 0, stream>>>(x, xt);
    pad_x    <<<dim3(BATCH * HH),      256, 0, stream>>>(xt);
    prep_w   <<<dim3(C_OUT, BATCH),    256, 0, stream>>>(mod, kmod, weights, wt);
    conv_wmma<<<dim3(WW / 32, HH, BATCH), 256, 0, stream>>>(xt, wt, out);
    norm_silu<<<dim3(HH, BATCH),       256, 0, stream>>>(out, gamma);
}
